// TripletLoss_12386685682268
// MI455X (gfx1250) — compile-verified, matrix operand order checked by reading, not run on hardware
//
#include <hip/hip_runtime.h>
#include <hip/hip_bf16.h>
#include <math.h>

typedef float  v2f __attribute__((ext_vector_type(2)));
typedef float  v8f __attribute__((ext_vector_type(8)));
typedef int    v8i __attribute__((ext_vector_type(8)));

#define NROWS   4096
#define DIM     512
#define LDIM    64
#define TMARGIN 0.2f
#define NCHUNK  16
#define TPC     (256 / NCHUNK)   // 16 column tiles per chunk
#define LDA     516              // padded LDS row stride (floats): 4*m banks distinct
#define LDL     68               // padded label row stride (bytes): 17*m banks distinct

// ---------------------------------------------------------------------------
// Kernel 1: row norms, label pack (int32 0/1 -> u8), inverse label norms,
//           (re)init hard-pos/neg sentinel arrays (bit patterns of -inf/+inf).
// ---------------------------------------------------------------------------
__global__ __launch_bounds__(256) void tl_prep(const float* __restrict__ X,
                                               const int* __restrict__ labels,
                                               float* __restrict__ sq,
                                               float* __restrict__ invn,
                                               unsigned char* __restrict__ pl,
                                               int* __restrict__ hp,
                                               unsigned* __restrict__ hn)
{
    int i = blockIdx.x * blockDim.x + threadIdx.x;
    if (i >= NROWS) return;

    const float* xr = X + (size_t)i * DIM;
    float s = 0.f;
    #pragma unroll 8
    for (int k = 0; k < DIM; ++k) { float v = xr[k]; s += v * v; }
    sq[i] = s;

    int cnt = 0;
    const int* lr = labels + (size_t)i * LDIM;
    unsigned char* pr = pl + (size_t)i * LDIM;
    #pragma unroll
    for (int k = 0; k < LDIM; ++k) { int v = lr[k]; pr[k] = (unsigned char)v; cnt += v; }
    float nrm = fmaxf(sqrtf((float)cnt), 1e-12f);
    invn[i] = 1.f / nrm;

    hp[i] = __float_as_int(-INFINITY);   // hard-pos accumulator (max, signed-int order ok: cands >= 0)
    hn[i] = (int)__float_as_uint(INFINITY); // hard-neg accumulator (min, unsigned order ok: cands >= 0)
}

// ---------------------------------------------------------------------------
// Kernel 2: fused Gram(WMMA f32 16x16x4) + label-Gram(WMMA i32 16x16x64 iu8)
//           + distance + mask + row-wise hard mining. One wave per
//           (16-row tile, 256-col chunk).
// ---------------------------------------------------------------------------
__global__ __launch_bounds__(32) void tl_mine(const float* __restrict__ X,
                                              const unsigned char* __restrict__ pl,
                                              const float* __restrict__ sq,
                                              const float* __restrict__ invn,
                                              int* __restrict__ hp,
                                              unsigned* __restrict__ hn)
{
    __shared__ float        ldsA[16 * LDA];
    __shared__ unsigned char ldsL[16 * LDL];

    const int lane    = threadIdx.x;
    const int half    = lane >> 4;        // 0: lanes 0-15, 1: lanes 16-31
    const int n16     = lane & 15;
    const int rowbase = blockIdx.x * 16;
    const int chunk   = blockIdx.y;

    // Stage 16 x 512 f32 A-strip into LDS (float4 granularity).
    for (int t = lane; t < 16 * (DIM / 4); t += 32) {
        int r  = t >> 7;                  // / (DIM/4)
        int kq = t & 127;
        float4 v = *(const float4*)(X + (size_t)(rowbase + r) * DIM + kq * 4);
        *(float4*)(&ldsA[r * LDA + kq * 4]) = v;
    }
    // Stage 16 x 64 u8 label rows.
    for (int t = lane; t < 16 * (LDIM / 4); t += 32) {
        int r  = t >> 4;
        int kd = t & 15;
        *(unsigned*)(&ldsL[r * LDL + kd * 4]) =
            *(const unsigned*)(pl + (size_t)(rowbase + r) * LDIM + kd * 4);
    }
    __syncthreads();

    // Fixed IU8 A-fragment (A 16x64 u8 layout: row = lane%16, dword i maps to
    // K = (i>>2)*32 + ((i&3)>>1)*16 + (i&1)*4 + 8*half).
    v8i la;
    #pragma unroll
    for (int i = 0; i < 8; ++i) {
        int j  = i & 3;
        int kk = ((i >> 2) << 5) + ((j >> 1) << 4) + ((j & 1) << 2) + half * 8;
        la[i] = *(const int*)(&ldsL[n16 * LDL + kk]);
    }

    // Per-wave row constants (C/D layout: VGPR v holds M = v + 8*half).
    float sqr[8], invr[8];
    #pragma unroll
    for (int v = 0; v < 8; ++v) {
        int r   = rowbase + v + 8 * half;
        sqr[v]  = sq[r];
        invr[v] = invn[r];
    }

    float runmax[8], runmin[8];
    #pragma unroll
    for (int v = 0; v < 8; ++v) { runmax[v] = -INFINITY; runmin[v] = INFINITY; }

    for (int jt = 0; jt < TPC; ++jt) {
        const int    colbase = (chunk * TPC + jt) * 16;
        const int    c       = colbase + n16;
        const float* Bcol    = X + (size_t)c * DIM;
        if (jt + 1 < TPC)
            __builtin_prefetch(X + (size_t)(c + 16) * DIM, 0, 1);  // global_prefetch_b8

        // G tile: 128 x V_WMMA_F32_16X16X4_F32, 4 accumulators for ILP.
        // A frag: lane reads A[m=n16][k + 2*half + {0,1}] ; B frag mirrored.
        v8f a0 = {}, a1 = {}, a2 = {}, a3 = {};
        const float* Arow = &ldsA[n16 * LDA + 2 * half];
        const float* Bc   = Bcol + 2 * half;
        #pragma unroll 8
        for (int k = 0; k < DIM; k += 16) {
            v2f aA = *(const v2f*)(Arow + k);
            v2f bA = *(const v2f*)(Bc + k);
            a0 = __builtin_amdgcn_wmma_f32_16x16x4_f32(false, aA, false, bA, (short)0, a0, false, false);
            v2f aB = *(const v2f*)(Arow + k + 4);
            v2f bB = *(const v2f*)(Bc + k + 4);
            a1 = __builtin_amdgcn_wmma_f32_16x16x4_f32(false, aB, false, bB, (short)0, a1, false, false);
            v2f aC = *(const v2f*)(Arow + k + 8);
            v2f bC = *(const v2f*)(Bc + k + 8);
            a2 = __builtin_amdgcn_wmma_f32_16x16x4_f32(false, aC, false, bC, (short)0, a2, false, false);
            v2f aD = *(const v2f*)(Arow + k + 12);
            v2f bD = *(const v2f*)(Bc + k + 12);
            a3 = __builtin_amdgcn_wmma_f32_16x16x4_f32(false, aD, false, bD, (short)0, a3, false, false);
        }

        // Exact label dot products: one V_WMMA_I32_16X16X64_IU8 (K == LDIM == 64).
        v8i lb;
        const unsigned char* Lc = pl + (size_t)c * LDIM;
        #pragma unroll
        for (int i = 0; i < 8; ++i) {
            int j  = i & 3;
            int kk = ((i >> 2) << 5) + ((j >> 1) << 4) + ((j & 1) << 2) + half * 8;
            lb[i] = *(const int*)(Lc + kk);
        }
        v8i zero = {};
        v8i dot = __builtin_amdgcn_wmma_i32_16x16x64_iu8(false, la, false, lb, zero, false, false);

        const float sqc  = sq[c];
        const float invc = invn[c];
        #pragma unroll
        for (int v = 0; v < 8; ++v) {
            int   r   = rowbase + v + 8 * half;
            float g   = (a0[v] + a1[v]) + (a2[v] + a3[v]);
            float sqd = sqr[v] + sqc - 2.f * g;
            float d   = (sqd > 0.f) ? sqrtf(sqd) : 0.f;
            float sim = (float)dot[v] * invr[v] * invc;
            if (r != c) {
                if (sim > 0.5f) runmax[v] = fmaxf(runmax[v], d);
                if (sim < 0.3f) runmin[v] = fminf(runmin[v], d);
            }
        }
    }

    // Butterfly reduce across the 16 lanes sharing each row (xor < 16 keeps halves separate).
    #pragma unroll
    for (int m = 8; m >= 1; m >>= 1) {
        #pragma unroll
        for (int v = 0; v < 8; ++v) {
            runmax[v] = fmaxf(runmax[v], __shfl_xor(runmax[v], m, 32));
            runmin[v] = fminf(runmin[v], __shfl_xor(runmin[v], m, 32));
        }
    }
    if (n16 == 0) {
        #pragma unroll
        for (int v = 0; v < 8; ++v) {
            int r = rowbase + v + 8 * half;
            atomicMax(&hp[r], __float_as_int(runmax[v]));       // candidates >= 0 or -inf sentinel
            atomicMin(&hn[r], __float_as_uint(runmin[v]));      // candidates >= 0 or +inf sentinel
        }
    }
}

// ---------------------------------------------------------------------------
// Kernel 3: masked mean of relu(hard_pos - hard_neg + margin) over valid rows.
// ---------------------------------------------------------------------------
__global__ __launch_bounds__(1024) void tl_finalize(const int* __restrict__ hp,
                                                    const unsigned* __restrict__ hn,
                                                    float* __restrict__ out)
{
    __shared__ float ssum[1024];
    __shared__ float scnt[1024];
    const int tid = threadIdx.x;
    float s = 0.f, c = 0.f;
    for (int r = tid; r < NROWS; r += 1024) {
        float p = __int_as_float(hp[r]);
        float n = __uint_as_float(hn[r]);
        if (p > -INFINITY && n < INFINITY) {          // has_pos && has_neg
            float l = p - n + TMARGIN;
            s += (l > 0.f) ? l : 0.f;
            c += 1.f;
        }
    }
    ssum[tid] = s; scnt[tid] = c;
    __syncthreads();
    for (int off = 512; off > 0; off >>= 1) {
        if (tid < off) { ssum[tid] += ssum[tid + off]; scnt[tid] += scnt[tid + off]; }
        __syncthreads();
    }
    if (tid == 0) out[0] = ssum[0] / fmaxf(scnt[0], 1.f);
}

extern "C" void kernel_launch(void* const* d_in, const int* in_sizes, int n_in,
                              void* d_out, int out_size, void* d_ws, size_t ws_size,
                              hipStream_t stream) {
    const float* X      = (const float*)d_in[0];
    const int*   labels = (const int*)d_in[1];
    float*       out    = (float*)d_out;

    // Workspace layout (~320 KB): sq | invn | hp | hn | packed labels
    float*         sq   = (float*)d_ws;
    float*         invn = sq + NROWS;
    int*           hp   = (int*)(invn + NROWS);
    unsigned*      hn   = (unsigned*)(hp + NROWS);
    unsigned char* pl   = (unsigned char*)(hn + NROWS);

    tl_prep<<<NROWS / 256, 256, 0, stream>>>(X, labels, sq, invn, pl, hp, hn);

    dim3 grid(NROWS / 16, NCHUNK);
    tl_mine<<<grid, 32, 0, stream>>>(X, pl, sq, invn, hp, hn);

    tl_finalize<<<1, 1024, 0, stream>>>(hp, hn, out);
}
